// MyMultiHeadAttention_66975720014198
// MI455X (gfx1250) — matrix operand order repaired and checked by reading, compiled, not run on hardware
//
#include <hip/hip_runtime.h>
#include <hip/hip_bf16.h>

// ---------------------------------------------------------------------------
// MHA forward for MI455X (gfx1250, wave32, WMMA + async global->LDS DMA).
// cvt(X)->bf16 ; pack weights to WMMA-B layout ; QKV GEMM (LDS-staged B) ;
// flash attention (LDS-staged K/V tiles) ; out proj + bias + residual.
// Matrix math: v_wmma_f32_16x16x32_bf16.  Staging: global_load_async_to_lds_b128
// double-buffered on ASYNCcnt so DMA of step k+1 overlaps WMMA of step k.
// ---------------------------------------------------------------------------

#define DEV __device__ __forceinline__

typedef __attribute__((ext_vector_type(16))) __bf16        v16bf;
typedef __attribute__((ext_vector_type(8)))  float         v8f;
typedef __attribute__((ext_vector_type(4)))  unsigned int  v4u;

static constexpr int B_ = 4, S_ = 1024, D_ = 1024, H_ = 16, DK_ = 64;

union Frag {            // 32 bytes, one WMMA A or B operand per lane
  v16bf          v;
  v4u            q[2];
  unsigned short u[16];
};

DEV unsigned short f2bf(float f) {            // round-to-nearest-even bf16
  union { float f; unsigned int u; } c; c.f = f;
  unsigned int u = c.u;
  u += 0x7fffu + ((u >> 16) & 1u);
  return (unsigned short)(u >> 16);
}

DEV v8f wmma_bf16(const Frag& a, const Frag& b, v8f c) {
  return __builtin_amdgcn_wmma_f32_16x16x32_bf16(
      false, a.v, false, b.v, (short)0, c, false, false);
}

// Async DMA: 16B per lane, global -> LDS.  LDS aperture puts the wave-relative
// LDS byte offset in addr[31:0], so truncating the generic pointer gives VDST.
DEV void async_g2l_b128(unsigned short* lds_dst, const unsigned short* gsrc) {
  const unsigned loff = (unsigned)(unsigned long long)lds_dst;
  asm volatile("global_load_async_to_lds_b128 %0, %1, off"
               :: "v"(loff), "v"(gsrc)
               : "memory");
}
#define S_WAIT_ASYNC_0() asm volatile("s_wait_asynccnt 0x0" ::: "memory")
#define S_WAIT_ASYNC_1() asm volatile("s_wait_asynccnt 0x1" ::: "memory")
#define S_WAIT_ASYNC_2() asm volatile("s_wait_asynccnt 0x2" ::: "memory")
#define S_WAIT_ASYNC_3() asm volatile("s_wait_asynccnt 0x3" ::: "memory")

// ---------------------------------------------------------------------------
// K0: fp32 -> bf16 elementwise
// ---------------------------------------------------------------------------
__global__ void k_f32_to_bf16(const float* __restrict__ in,
                              unsigned short* __restrict__ out, int n) {
  int i = blockIdx.x * blockDim.x + threadIdx.x;
  if (i < n) out[i] = f2bf(in[i]);
}

// ---------------------------------------------------------------------------
// K1: pack (batch, D, N) fp32 weight into WMMA B-fragment order:
//     out[batch][D/32][N/16][lane 0..31][e 0..15]  (bf16)
// B 32x16 16-bit layout: col n = lane&15, K = (e>>3)*16 + (lane>>4)*8 + (e&7)
// ---------------------------------------------------------------------------
__global__ void k_pack_b(const float* __restrict__ W, unsigned short* __restrict__ out,
                         int D, int N, long long total) {
  long long t = (long long)blockIdx.x * blockDim.x + threadIdx.x;
  if (t >= total) return;
  const int e    = (int)(t & 15);
  const int lane = (int)((t >> 4) & 31);
  long long r    = t >> 9;
  const int ntiles = N >> 4;
  const int nt = (int)(r % ntiles); r /= ntiles;
  const int ktiles = D >> 5;
  const int kt = (int)(r % ktiles);
  const long long bidx = r / ktiles;
  const int k = kt * 32 + ((e >> 3) << 4) + (lane >> 4) * 8 + (e & 7);
  const int n = nt * 16 + (lane & 15);
  out[t] = f2bf(W[(bidx * D + k) * (long long)N + n]);
}

// ---------------------------------------------------------------------------
// K2: QKV projection.  Block = 8 waves = 128 rows of one (b,h).
// Weight B-frags (shared by all waves) staged via async DMA, double-buffered:
// 12 KB/step = 3 async b128 per wave.  Q pre-scaled by 1/8; V stored
// transposed [B,H,64,S].
// ---------------------------------------------------------------------------
__global__ __launch_bounds__(256) void k_qkv(
    const unsigned short* __restrict__ Xb,
    const unsigned short* __restrict__ Wqp,
    const unsigned short* __restrict__ Wkp,
    const unsigned short* __restrict__ Wvp,
    const float* __restrict__ bq, const float* __restrict__ bk,
    const float* __restrict__ bv,
    unsigned short* __restrict__ Q, unsigned short* __restrict__ Kc,
    unsigned short* __restrict__ Vt) {
  __shared__ __align__(16) unsigned short wbuf[2][3 * 2048];   // 24 KB
  const int t    = threadIdx.x;
  const int lane = t & 31;
  const int wv   = t >> 5;
  const int idx  = lane & 15;
  const int hh   = lane >> 4;
  const int rb   = blockIdx.x & 7;          // S/128 row blocks
  const int h    = (blockIdx.x >> 3) & 15;
  const int b    = blockIdx.x >> 7;
  const int i0   = rb * 128 + wv * 16;

  const size_t whead = (size_t)h * (D_ >> 5) * 2048;
  const unsigned short* qsrc = Wqp + whead;
  const unsigned short* ksrc = Wkp + whead;
  const unsigned short* vsrc = Wvp + whead;
  const int co = t * 8;                      // one 16B chunk per thread

  // prologue: stage kt = 0
  async_g2l_b128(&wbuf[0][co],        qsrc + co);
  async_g2l_b128(&wbuf[0][2048 + co], ksrc + co);
  async_g2l_b128(&wbuf[0][4096 + co], vsrc + co);

  v8f aq[4] = {}, ak[4] = {}, av[4] = {};
  const unsigned short* xrow = Xb + ((size_t)b * S_ + (i0 + idx)) * D_ + hh * 8;

  for (int kt = 0; kt < (D_ >> 5); ++kt) {
    const int cur = kt & 1;
    if (kt + 1 < (D_ >> 5)) {
      const size_t s = (size_t)(kt + 1) * 2048;
      async_g2l_b128(&wbuf[cur ^ 1][co],        qsrc + s + co);
      async_g2l_b128(&wbuf[cur ^ 1][2048 + co], ksrc + s + co);
      async_g2l_b128(&wbuf[cur ^ 1][4096 + co], vsrc + s + co);
      S_WAIT_ASYNC_3();          // in-order: current buffer's 3 DMAs done
    } else {
      S_WAIT_ASYNC_0();
    }
    __syncthreads();

    Frag a;                                  // A 16x32: row = lane&15
    a.q[0] = *(const v4u*)(xrow + kt * 32);          // K = hh*8 + 0..7
    a.q[1] = *(const v4u*)(xrow + kt * 32 + 16);     // K = 16 + hh*8 + 0..7
    if (kt + 2 < (D_ >> 5)) __builtin_prefetch(xrow + kt * 32 + 64, 0, 0);

    const unsigned short* wb = &wbuf[cur][lane * 16];
#pragma unroll
    for (int nt = 0; nt < 4; ++nt) {
      Frag bqf, bkf, bvf;
      bqf.q[0] = *(const v4u*)(wb + nt * 512);
      bqf.q[1] = *(const v4u*)(wb + nt * 512 + 8);
      bkf.q[0] = *(const v4u*)(wb + 2048 + nt * 512);
      bkf.q[1] = *(const v4u*)(wb + 2048 + nt * 512 + 8);
      bvf.q[0] = *(const v4u*)(wb + 4096 + nt * 512);
      bvf.q[1] = *(const v4u*)(wb + 4096 + nt * 512 + 8);
      aq[nt] = wmma_bf16(a, bqf, aq[nt]);
      ak[nt] = wmma_bf16(a, bkf, ak[nt]);
      av[nt] = wmma_bf16(a, bvf, av[nt]);
    }
    __syncthreads();             // reads done before next DMA overwrites
  }

  // C layout: element r of lane (hh,idx) = row r+8*hh, col idx
  const size_t bh = (size_t)b * H_ + h;
#pragma unroll
  for (int nt = 0; nt < 4; ++nt) {
    const int n = nt * 16 + idx;
    const float bqv = bq[h * DK_ + n];
    const float bkv = bk[h * DK_ + n];
    const float bvv = bv[h * DK_ + n];
#pragma unroll
    for (int r = 0; r < 8; ++r) {
      const int srow = i0 + r + 8 * hh;
      Q [(bh * S_ + srow) * DK_ + n] = f2bf((aq[nt][r] + bqv) * 0.125f);
      Kc[(bh * S_ + srow) * DK_ + n] = f2bf(ak[nt][r] + bkv);
      Vt[(bh * DK_ + n) * S_ + srow] = f2bf(av[nt][r] + bvv);   // transposed
    }
  }
}

// ---------------------------------------------------------------------------
// K3: flash attention.  Block = 8 waves = 128 query rows of one (b,h).
// K tile (32x64) and V^T tile (64x32) shared by all waves -> async DMA into
// LDS, double-buffered (2 async b128 per wave per 32-key step).  Scores and
// PV on WMMA; online softmax via shfl_xor in 16-lane halves; P goes through
// per-wave LDS for the C->A layout change.
// ---------------------------------------------------------------------------
__global__ __launch_bounds__(256) void k_attn(
    const unsigned short* __restrict__ Q,
    const unsigned short* __restrict__ Kc,
    const unsigned short* __restrict__ Vt,
    unsigned short* __restrict__ Zc) {
  __shared__ __align__(16) unsigned short psh[8][16][32];    // 8 KB, per-wave P
  __shared__ __align__(16) unsigned short ktile[2][2048];    // 8 KB
  __shared__ __align__(16) unsigned short vtile[2][2048];    // 8 KB
  const int t    = threadIdx.x;
  const int lane = t & 31;
  const int wv   = t >> 5;
  const int idx  = lane & 15;
  const int hh   = lane >> 4;
  const int rb   = blockIdx.x & 7;
  const int h    = (blockIdx.x >> 3) & 15;
  const int b    = blockIdx.x >> 7;
  const int i0   = rb * 128 + wv * 16;
  const size_t bh = (size_t)b * H_ + h;

  const unsigned short* kbase = Kc + bh * (size_t)S_ * DK_;
  const unsigned short* vbase = Vt + bh * (size_t)DK_ * S_;

  // copy plan: K tile is 2048 contiguous shorts; V tile gathers 64 rows x 64B
  const int kco  = t * 8;
  const int vr   = t >> 2;                 // V row 0..63
  const int vs   = (t & 3) * 8;            // 16B segment within 64B row chunk

  // prologue: stage j0 = 0
  async_g2l_b128(&ktile[0][kco], kbase + kco);
  async_g2l_b128(&vtile[0][vr * 32 + vs], vbase + (size_t)vr * S_ + vs);

  Frag qf[2];
  {
    const unsigned short* qrow = Q + (bh * S_ + (i0 + idx)) * DK_ + hh * 16;
    qf[0].q[0] = *(const v4u*)(qrow);
    qf[0].q[1] = *(const v4u*)(qrow + 8);
    qf[1].q[0] = *(const v4u*)(qrow + 32);
    qf[1].q[1] = *(const v4u*)(qrow + 40);
  }

  float m[8], l[8];
  v8f o[4] = {};
#pragma unroll
  for (int r = 0; r < 8; ++r) { m[r] = -3.0e38f; l[r] = 0.f; }

  for (int j0 = 0; j0 < S_; j0 += 32) {
    const int cur = (j0 >> 5) & 1;
    if (j0 + 32 < S_) {
      async_g2l_b128(&ktile[cur ^ 1][kco],
                     kbase + (size_t)(j0 + 32) * DK_ + kco);
      async_g2l_b128(&vtile[cur ^ 1][vr * 32 + vs],
                     vbase + (size_t)vr * S_ + (j0 + 32) + vs);
      S_WAIT_ASYNC_2();
    } else {
      S_WAIT_ASYNC_0();
    }
    __syncthreads();

    // --- scores: two 16x16 tiles over 32 keys (K frags from LDS) ---
    v8f st[2];
#pragma unroll
    for (int jt = 0; jt < 2; ++jt) {
      const unsigned short* krow = &ktile[cur][(jt * 16 + idx) * 64 + hh * 16];
      Frag kf0, kf1;
      kf0.q[0] = *(const v4u*)(krow);        // d_k = hh*16 + 0..15
      kf0.q[1] = *(const v4u*)(krow + 8);
      kf1.q[0] = *(const v4u*)(krow + 32);   // d_k = 32 + hh*16 + 0..15
      kf1.q[1] = *(const v4u*)(krow + 40);
      v8f s = {};
      s = wmma_bf16(qf[0], kf0, s);
      s = wmma_bf16(qf[1], kf1, s);
      st[jt] = s;
    }
    // --- online softmax (rows live across 16-lane halves) ---
    float nm[8], sc[8];
#pragma unroll
    for (int r = 0; r < 8; ++r) {
      float v = fmaxf(st[0][r], st[1][r]);
#pragma unroll
      for (int d = 1; d < 16; d <<= 1) v = fmaxf(v, __shfl_xor(v, d, 32));
      nm[r] = fmaxf(m[r], v);
      sc[r] = __expf(m[r] - nm[r]);
      m[r]  = nm[r];
    }
#pragma unroll
    for (int jt = 0; jt < 2; ++jt)
#pragma unroll
      for (int r = 0; r < 8; ++r) st[jt][r] = __expf(st[jt][r] - nm[r]);
#pragma unroll
    for (int r = 0; r < 8; ++r) {
      float v = st[0][r] + st[1][r];
#pragma unroll
      for (int d = 1; d < 16; d <<= 1) v += __shfl_xor(v, d, 32);
      l[r] = l[r] * sc[r] + v;
    }
#pragma unroll
    for (int nt = 0; nt < 4; ++nt)
#pragma unroll
      for (int r = 0; r < 8; ++r) o[nt][r] *= sc[r];
    // --- P: C-layout -> A-layout via per-wave LDS ---
#pragma unroll
    for (int jt = 0; jt < 2; ++jt)
#pragma unroll
      for (int r = 0; r < 8; ++r)
        psh[wv][r + 8 * hh][jt * 16 + idx] = f2bf(st[jt][r]);
    __builtin_amdgcn_wave_barrier();
    Frag pf;
    pf.q[0] = *(const v4u*)&psh[wv][idx][hh * 16];
    pf.q[1] = *(const v4u*)&psh[wv][idx][hh * 16 + 8];
    __builtin_amdgcn_wave_barrier();
    // --- O += P x V  (V frags from LDS, B[kk][n] = Vt[n][j0+kk]) ---
#pragma unroll
    for (int nt = 0; nt < 4; ++nt) {
      const unsigned short* vrow = &vtile[cur][(nt * 16 + idx) * 32 + hh * 16];
      Frag vf;
      vf.q[0] = *(const v4u*)(vrow);
      vf.q[1] = *(const v4u*)(vrow + 8);
      o[nt] = wmma_bf16(pf, vf, o[nt]);
    }
    __syncthreads();             // tile reads done before next DMA overwrites
  }
  // --- normalize, store concat-head Z (bf16) ---
  float inv[8];
#pragma unroll
  for (int r = 0; r < 8; ++r) inv[r] = 1.0f / l[r];
#pragma unroll
  for (int nt = 0; nt < 4; ++nt) {
    const int n = h * DK_ + nt * 16 + idx;
#pragma unroll
    for (int r = 0; r < 8; ++r) {
      const int srow = i0 + r + 8 * hh;
      Zc[((size_t)b * S_ + srow) * (H_ * DK_) + n] = f2bf(o[nt][r] * inv[r]);
    }
  }
}

// ---------------------------------------------------------------------------
// K4: output projection  out = Zc @ Wo + bo + X   (f32 out)
// M=4096, K=1024, N=1024.  Wo frags async-staged to LDS (4 KB/step).
// ---------------------------------------------------------------------------
__global__ __launch_bounds__(256) void k_proj(
    const unsigned short* __restrict__ Zc,
    const unsigned short* __restrict__ Wop,
    const float* __restrict__ bo,
    const float* __restrict__ X,
    float* __restrict__ out) {
  __shared__ __align__(16) unsigned short wbuf[2][2048];   // 8 KB
  const int t    = threadIdx.x;
  const int lane = t & 31;
  const int wv   = t >> 5;
  const int idx  = lane & 15;
  const int hh   = lane >> 4;
  const int cb   = blockIdx.x & 15;     // 1024/64 col blocks
  const int rb   = blockIdx.x >> 4;     // 4096/128 row blocks
  const int i0   = rb * 128 + wv * 16;
  const int n0   = cb * 64;

  const unsigned short* wsrc = Wop + (size_t)(n0 >> 4) * 512;
  const int co = t * 8;
  async_g2l_b128(&wbuf[0][co], wsrc + co);          // stage kt = 0

  v8f acc[4] = {};
  const unsigned short* zrow = Zc + (size_t)(i0 + idx) * D_ + hh * 8;

  for (int kt = 0; kt < (D_ >> 5); ++kt) {
    const int cur = kt & 1;
    if (kt + 1 < (D_ >> 5)) {
      async_g2l_b128(&wbuf[cur ^ 1][co],
                     wsrc + (size_t)(kt + 1) * 32768 + co);
      S_WAIT_ASYNC_1();
    } else {
      S_WAIT_ASYNC_0();
    }
    __syncthreads();

    Frag a;
    a.q[0] = *(const v4u*)(zrow + kt * 32);
    a.q[1] = *(const v4u*)(zrow + kt * 32 + 16);
    if (kt + 2 < (D_ >> 5)) __builtin_prefetch(zrow + kt * 32 + 64, 0, 0);

    const unsigned short* wb = &wbuf[cur][lane * 16];
#pragma unroll
    for (int nt = 0; nt < 4; ++nt) {
      Frag bf;
      bf.q[0] = *(const v4u*)(wb + nt * 512);
      bf.q[1] = *(const v4u*)(wb + nt * 512 + 8);
      acc[nt] = wmma_bf16(a, bf, acc[nt]);
    }
    __syncthreads();
  }
#pragma unroll
  for (int nt = 0; nt < 4; ++nt) {
    const int n = n0 + nt * 16 + idx;
    const float bias = bo[n];
#pragma unroll
    for (int r = 0; r < 8; ++r) {
      const size_t row = (size_t)(i0 + r + 8 * hh);
      out[row * D_ + n] = acc[nt][r] + bias + X[row * D_ + n];
    }
  }
}

// ---------------------------------------------------------------------------
// Host-side launcher
// ---------------------------------------------------------------------------
extern "C" void kernel_launch(void* const* d_in, const int* in_sizes, int n_in,
                              void* d_out, int out_size, void* d_ws, size_t ws_size,
                              hipStream_t stream) {
  (void)in_sizes; (void)n_in; (void)out_size; (void)ws_size;
  const float* X  = (const float*)d_in[0];
  const float* Wk = (const float*)d_in[1];
  const float* bk = (const float*)d_in[2];
  const float* Wq = (const float*)d_in[3];
  const float* bq = (const float*)d_in[4];
  const float* Wv = (const float*)d_in[5];
  const float* bv = (const float*)d_in[6];
  const float* Wo = (const float*)d_in[7];
  const float* bo = (const float*)d_in[8];
  float* out = (float*)d_out;

  const size_t nX  = (size_t)B_ * S_ * D_;       // 4M
  const size_t nW  = (size_t)H_ * D_ * DK_;      // 1M per QKV weight
  const size_t nWo = (size_t)D_ * D_;            // 1M
  const size_t nQ  = (size_t)B_ * H_ * S_ * DK_; // 4M

  unsigned short* p   = (unsigned short*)d_ws;   // 48 MB total bf16 scratch
  unsigned short* Xb  = p; p += nX;
  unsigned short* Wqp = p; p += nW;
  unsigned short* Wkp = p; p += nW;
  unsigned short* Wvp = p; p += nW;
  unsigned short* Wop = p; p += nWo;
  unsigned short* Qb  = p; p += nQ;
  unsigned short* Kb  = p; p += nQ;
  unsigned short* Vtb = p; p += nQ;
  unsigned short* Zc  = p; p += nX;

  k_f32_to_bf16<<<(int)(nX / 256), 256, 0, stream>>>(X, Xb, (int)nX);
  k_pack_b<<<(int)(nW / 256), 256, 0, stream>>>(Wq, Wqp, D_, DK_, (long long)nW);
  k_pack_b<<<(int)(nW / 256), 256, 0, stream>>>(Wk, Wkp, D_, DK_, (long long)nW);
  k_pack_b<<<(int)(nW / 256), 256, 0, stream>>>(Wv, Wvp, D_, DK_, (long long)nW);
  k_pack_b<<<(int)(nWo / 256), 256, 0, stream>>>(Wo, Wop, D_, D_, (long long)nWo);

  k_qkv<<<B_ * H_ * (S_ / 128), 256, 0, stream>>>(Xb, Wqp, Wkp, Wvp,
                                                  bq, bk, bv, Qb, Kb, Vtb);
  k_attn<<<B_ * H_ * (S_ / 128), 256, 0, stream>>>(Qb, Kb, Vtb, Zc);
  k_proj<<<(B_ * S_ / 128) * (D_ / 64), 256, 0, stream>>>(Zc, Wop, bo, X, out);
}
